// BaselineT_13486197309988
// MI455X (gfx1250) — compile-verified, matrix operand order checked by reading, not run on hardware
//
#include <hip/hip_runtime.h>

// ---------------------------------------------------------------------------
// Types for CDNA5 (gfx1250) WMMA: wave32, V_WMMA_F32_16X16X32_BF16
// ---------------------------------------------------------------------------
typedef __bf16 bf16;
typedef __attribute__((ext_vector_type(16))) __bf16 v16bf;
typedef __attribute__((ext_vector_type(8)))  __bf16 v8bf;
typedef __attribute__((ext_vector_type(8)))  float  v8f;

#define WMMA_BF16(a, b, c) \
    __builtin_amdgcn_wmma_f32_16x16x32_bf16(false, (a), false, (b), (short)0, (c), false, false)

// Model dims
constexpr int Dm   = 768;
constexpr int DFm  = 3072;
constexpr int Hn   = 12;
constexpr int HDn  = 64;
constexpr int Ln   = 12;
constexpr int Vn   = 32000;
constexpr int Bn   = 2;
constexpr int Tn   = 2048;
constexpr int Mrows = Bn * Tn;      // 4096
constexpr int QKVn  = 3 * Hn * HDn; // 2304

// ---------------------------------------------------------------------------
// Helpers
// ---------------------------------------------------------------------------
__device__ __forceinline__ bf16 f2bf(float f) {
    unsigned u = __float_as_uint(f);
    u += 0x7FFFu + ((u >> 16) & 1u);           // round-to-nearest-even
    unsigned short h = (unsigned short)(u >> 16);
    return __builtin_bit_cast(bf16, h);
}

// Load one wave32 WMMA fragment (A- or B-matrix, bf16, K=32 step).
// Lane's data = 8 contiguous bf16 at p, plus 8 at p+16 (caller adds half*8).
__device__ __forceinline__ v16bf load_frag(const bf16* p) {
    v8bf lo = *(const v8bf*)(p);
    v8bf hi = *(const v8bf*)(p + 16);
    return __builtin_shufflevector(lo, hi, 0,1,2,3,4,5,6,7,8,9,10,11,12,13,14,15);
}

__device__ __forceinline__ float rowmax16(float v) {
    #pragma unroll
    for (int off = 1; off < 16; off <<= 1) v = fmaxf(v, __shfl_xor(v, off, 32));
    return v;
}
__device__ __forceinline__ float rowsum16(float v) {
    #pragma unroll
    for (int off = 1; off < 16; off <<= 1) v += __shfl_xor(v, off, 32);
    return v;
}

// ---------------------------------------------------------------------------
// Tensor Data Mover: 2D tile (128 rows x 32 bf16) global -> LDS
// D# per CDNA5 ISA (count=1, type=2, data_size=2B)
// ---------------------------------------------------------------------------
#if __has_builtin(__builtin_amdgcn_tensor_load_to_lds)
#define HAVE_TDM 1
typedef __attribute__((ext_vector_type(4))) unsigned int u32x4;
typedef __attribute__((ext_vector_type(8))) int i32x8;
typedef __attribute__((ext_vector_type(4))) int i32x4;

__device__ __forceinline__ void tdm_load_tile(unsigned lds_byte_off, const void* gptr,
                                              unsigned row_stride_elems) {
    unsigned long long ga = (unsigned long long)gptr;
    u32x4 g0;
    g0.x = 1u;                                                  // count=1 (valid user D#)
    g0.y = lds_byte_off;                                        // lds_addr (bytes)
    g0.z = (unsigned)ga;                                        // global_addr[31:0]
    g0.w = (unsigned)((ga >> 32) & 0x01FFFFFFu) | (2u << 30);   // addr[56:32] | type=2
    i32x8 g1;
    g1[0] = (int)(1u << 16);                                    // data_size=1 -> 2 bytes
    g1[1] = (int)((row_stride_elems & 0xFFFFu) << 16);          // tensor_dim0 lo16
    g1[2] = (int)(((row_stride_elems >> 16) & 0xFFFFu)          // tensor_dim0 hi16
          |       (0xFFFFu << 16));                             // tensor_dim1 lo16 (large)
    g1[3] = (int)(0x000Fu | (32u << 16));                       // tensor_dim1 hi | tile_dim0=32
    g1[4] = (int)128;                                           // tile_dim1=128, tile_dim2=0
    g1[5] = (int)row_stride_elems;                              // tensor_dim0_stride lo32
    g1[6] = 0;                                                  // stride hi | dim1_stride lo
    g1[7] = 0;
    i32x4 z4 = {0, 0, 0, 0};
#if __clang_major__ >= 23
    i32x8 z8 = {0, 0, 0, 0, 0, 0, 0, 0};
    __builtin_amdgcn_tensor_load_to_lds(g0, g1, z4, z4, z8, 0);
#else
    __builtin_amdgcn_tensor_load_to_lds(g0, g1, z4, z4, 0);
#endif
}
#else
#define HAVE_TDM 0
#endif

// ---------------------------------------------------------------------------
// Embedding: x[b,t,:] = te[ids[b,t],:] + pe[t,:]
// ---------------------------------------------------------------------------
__global__ void embed_kernel(const int* __restrict__ ids, const float* __restrict__ te,
                             const float* __restrict__ pe, float* __restrict__ x) {
    size_t idx = (size_t)blockIdx.x * 256 + threadIdx.x;
    if (idx >= (size_t)Mrows * Dm) return;
    int d  = (int)(idx % Dm);
    int bt = (int)(idx / Dm);
    int t  = bt % Tn;
    x[idx] = te[(size_t)ids[bt] * Dm + d] + pe[(size_t)t * Dm + d];
}

// fp32 -> bf16 straight convert (tied embedding used as Wt [V,D])
__global__ void convert_kernel(const float* __restrict__ in, bf16* __restrict__ out, size_t n) {
    size_t idx = (size_t)blockIdx.x * 256 + threadIdx.x;
    if (idx < n) out[idx] = f2bf(in[idx]);
}

// fp32 W[K,N] -> bf16 Wt[N,K] (transpose + convert): B-fragments become K-contiguous
__global__ void convert_transpose_kernel(const float* __restrict__ W, bf16* __restrict__ Wt,
                                         int K, int N) {
    size_t idx = (size_t)blockIdx.x * 256 + threadIdx.x;
    if (idx >= (size_t)K * N) return;
    int n = (int)(idx % N);
    int k = (int)(idx / N);
    Wt[(size_t)n * K + k] = f2bf(W[idx]);
}

// Extract V from qkv and store transposed: vt[(b,h,hd), t]  (PV B-frags contiguous)
__global__ void vtrans_kernel(const bf16* __restrict__ qkvb, bf16* __restrict__ vt) {
    size_t idx = (size_t)blockIdx.x * 256 + threadIdx.x;
    if (idx >= (size_t)Mrows * Dm) return;
    int c  = (int)(idx % Dm);
    int bt = (int)(idx / Dm);
    int h  = c >> 6, hd = c & 63;
    int b  = bt / Tn, t = bt % Tn;
    vt[(((size_t)b * Hn + h) * HDn + hd) * Tn + t] = qkvb[(size_t)bt * QKVn + 2 * Hn * HDn + c];
}

// ---------------------------------------------------------------------------
// LayerNorm: one row (D=768) per 256-thread block; bf16 output (GEMM A input)
// ---------------------------------------------------------------------------
__global__ void __launch_bounds__(256)
layernorm_kernel(const float* __restrict__ x, const float* __restrict__ g,
                 const float* __restrict__ bi, bf16* __restrict__ out) {
    __shared__ float s1[256], s2[256];
    int tid = threadIdx.x;
    const float* xr = x + (size_t)blockIdx.x * Dm;
    float a = 0.f, b = 0.f;
    for (int i = tid; i < Dm; i += 256) { float v = xr[i]; a += v; b += v * v; }
    s1[tid] = a; s2[tid] = b;
    __syncthreads();
    for (int off = 128; off > 0; off >>= 1) {
        if (tid < off) { s1[tid] += s1[tid + off]; s2[tid] += s2[tid + off]; }
        __syncthreads();
    }
    float mu  = s1[0] * (1.f / Dm);
    float var = s2[0] * (1.f / Dm) - mu * mu;
    float r   = rsqrtf(var + 1e-5f);
    bf16* orow = out + (size_t)blockIdx.x * Dm;
    for (int i = tid; i < Dm; i += 256)
        orow[i] = f2bf((xr[i] - mu) * r * g[i] + bi[i]);
}

// ---------------------------------------------------------------------------
// WMMA GEMM: C[M,N] = A[M,K](bf16) x Wt[N,K]^T (+bias)(silu)(+res)
// 256 threads = 8 waves (2 M-groups x 4 N-groups); workgroup tile 128x128.
// Each wave: 64x32 output = 4x2 WMMA accumulators. A/B 32-wide k-slices are
// staged in LDS, double-buffered, fed by the Tensor Data Mover (wave 0 issues
// tensor_load_to_lds; s_wait_tensorcnt pipelines current vs next buffer).
// ---------------------------------------------------------------------------
template<bool OUT_BF16, bool SILU>
__global__ void __launch_bounds__(256)
gemm_wmma_kernel(const bf16* __restrict__ A, const bf16* __restrict__ Wt,
                 const float* __restrict__ bias, const float* __restrict__ res,
                 void* __restrict__ out, int M, int N, int K) {
    __shared__ bf16 smem[2][2][128 * 32];   // [buf][A|B][row*32+col], 32 KB
    (void)M;
    int lane = threadIdx.x & 31;
    int wave = threadIdx.x >> 5;
    int half = lane >> 4;
    int l16  = lane & 15;
    int mw = wave & 1;          // 0..1 -> 64-row group
    int nw = wave >> 1;         // 0..3 -> 32-col group
    int m_blk = blockIdx.y * 128;
    int n_blk = blockIdx.x * 128;

    const bf16* Ablk = A  + (size_t)m_blk * K;
    const bf16* Bblk = Wt + (size_t)n_blk * K;

    v8f acc[4][2];
    #pragma unroll
    for (int mt = 0; mt < 4; mt++)
        #pragma unroll
        for (int nt = 0; nt < 2; nt++) acc[mt][nt] = (v8f){};

    const int nsteps = K / 32;
#if HAVE_TDM
    if (wave == 0) {                 // prologue: fill buffer 0
        tdm_load_tile(0u,                 Ablk, (unsigned)K);
        tdm_load_tile(128u * 32u * 2u,    Bblk, (unsigned)K);
    }
#endif
    for (int s = 0; s < nsteps; s++) {
        int buf = s & 1;
        int k = s * 32;
#if HAVE_TDM
        if (wave == 0) {
            if (s + 1 < nsteps) {    // prefetch next k-slice into other buffer
                unsigned nb = (unsigned)((buf ^ 1) * 2 * 128 * 32 * 2);
                tdm_load_tile(nb,                  Ablk + k + 32, (unsigned)K);
                tdm_load_tile(nb + 128u * 32u * 2u, Bblk + k + 32, (unsigned)K);
                asm volatile("s_wait_tensorcnt 0x2" ::: "memory"); // current pair done
            } else {
                asm volatile("s_wait_tensorcnt 0x0" ::: "memory");
            }
        }
        __syncthreads();
#else
        {   // cooperative fallback: 256 threads copy both 8 KB tiles
            int r = threadIdx.x >> 1, c = (threadIdx.x & 1) * 16;
            *(v8bf*)(&smem[buf][0][r * 32 + c])     = *(const v8bf*)(Ablk + (size_t)r * K + k + c);
            *(v8bf*)(&smem[buf][0][r * 32 + c + 8]) = *(const v8bf*)(Ablk + (size_t)r * K + k + c + 8);
            *(v8bf*)(&smem[buf][1][r * 32 + c])     = *(const v8bf*)(Bblk + (size_t)r * K + k + c);
            *(v8bf*)(&smem[buf][1][r * 32 + c + 8]) = *(const v8bf*)(Bblk + (size_t)r * K + k + c + 8);
        }
        __syncthreads();
#endif
        const bf16* sa = smem[buf][0];
        const bf16* sb = smem[buf][1];
        v16bf af[4], bfr[2];
        #pragma unroll
        for (int mt = 0; mt < 4; mt++)
            af[mt] = load_frag(sa + (mw * 64 + mt * 16 + l16) * 32 + half * 8);
        #pragma unroll
        for (int nt = 0; nt < 2; nt++)
            bfr[nt] = load_frag(sb + (nw * 32 + nt * 16 + l16) * 32 + half * 8);
        #pragma unroll
        for (int mt = 0; mt < 4; mt++)
            #pragma unroll
            for (int nt = 0; nt < 2; nt++)
                acc[mt][nt] = WMMA_BF16(af[mt], bfr[nt], acc[mt][nt]);
        __syncthreads();             // all reads done before buf is re-filled
    }

    #pragma unroll
    for (int nt = 0; nt < 2; nt++) {
        int col = n_blk + nw * 32 + nt * 16 + l16;
        float bv = bias ? bias[col] : 0.f;
        #pragma unroll
        for (int mt = 0; mt < 4; mt++) {
            #pragma unroll
            for (int i = 0; i < 8; i++) {
                int row = m_blk + mw * 64 + mt * 16 + i + half * 8;
                float v = acc[mt][nt][i] + bv;
                if (SILU) v = v / (1.f + __expf(-v));
                if (res)  v += res[(size_t)row * N + col];
                if (OUT_BF16) ((bf16*)out)[(size_t)row * N + col] = f2bf(v);
                else          ((float*)out)[(size_t)row * N + col] = v;
            }
        }
    }
}

// ---------------------------------------------------------------------------
// Flash attention: one wave per 16-query tile; keys streamed 32 at a time.
// S=QK^T and O+=PV via WMMA; P re-laid out C-format -> A-format via LDS.
// Q,K from qkv [b,t,3*H*HD]; V from transposed vt[(b,h,hd), t].
// ---------------------------------------------------------------------------
__global__ void __launch_bounds__(128)
attention_kernel(const bf16* __restrict__ qkv, const bf16* __restrict__ vt,
                 bf16* __restrict__ o) {
    int lane = threadIdx.x & 31;
    int wave = threadIdx.x >> 5;
    int half = lane >> 4;
    int l16  = lane & 15;

    int qc = blockIdx.x % (Tn / 64);
    int hh = (blockIdx.x / (Tn / 64)) % Hn;
    int b  = blockIdx.x / ((Tn / 64) * Hn);
    int q0 = qc * 64 + wave * 16;

    __shared__ bf16 Pb[4][16 * 32];
    bf16* pb = Pb[wave];

    const size_t rs = QKVn;
    const bf16* qrow = qkv + ((size_t)(b * Tn + q0 + l16)) * rs + hh * HDn;
    v16bf qa0 = load_frag(qrow + half * 8);        // hd 0..31
    v16bf qa1 = load_frag(qrow + 32 + half * 8);   // hd 32..63

    const bf16* vbase = vt + (((size_t)b * Hn + hh) * HDn) * Tn;

    float mrun[8], lrun[8];
    v8f o0 = {}, o1 = {}, o2 = {}, o3 = {};
    #pragma unroll
    for (int i = 0; i < 8; i++) { mrun[i] = -1e30f; lrun[i] = 0.f; }

    for (int kk = 0; kk < q0 + 16; kk += 32) {
        const bf16* k0row = qkv + ((size_t)(b * Tn + kk + l16)) * rs + Hn * HDn + hh * HDn;
        const bf16* k1row = k0row + (size_t)16 * rs;
        v8f s0 = {}, s1 = {};
        v16bf kb;
        kb = load_frag(k0row + half * 8);       s0 = WMMA_BF16(qa0, kb, s0);
        kb = load_frag(k0row + 32 + half * 8);  s0 = WMMA_BF16(qa1, kb, s0);
        kb = load_frag(k1row + half * 8);       s1 = WMMA_BF16(qa0, kb, s1);
        kb = load_frag(k1row + 32 + half * 8);  s1 = WMMA_BF16(qa1, kb, s1);

        int key0 = kk + l16, key1 = kk + 16 + l16;
        #pragma unroll
        for (int i = 0; i < 8; i++) {
            int t = q0 + i + half * 8;
            float a  = s0[i] * 0.125f;   // 1/sqrt(64)
            float c2 = s1[i] * 0.125f;
            if (key0 > t) a  = -1e30f;   // causal mask
            if (key1 > t) c2 = -1e30f;
            float m    = rowmax16(fmaxf(a, c2));
            float mn   = fmaxf(mrun[i], m);
            float corr = __expf(mrun[i] - mn);
            float p0   = __expf(a - mn);
            float p1   = __expf(c2 - mn);
            lrun[i] = lrun[i] * corr + rowsum16(p0 + p1);
            mrun[i] = mn;
            o0[i] *= corr; o1[i] *= corr; o2[i] *= corr; o3[i] *= corr;
            int r = i + half * 8;
            pb[r * 32 + l16]      = f2bf(p0);
            pb[r * 32 + 16 + l16] = f2bf(p1);
        }
        asm volatile("s_wait_dscnt 0x0" ::: "memory");  // LDS write -> read fence
        v16bf pf = load_frag(pb + l16 * 32 + half * 8); // P as A-fragment (16x32)

        #pragma unroll
        for (int nt = 0; nt < 4; nt++) {
            v16bf vf = load_frag(vbase + (size_t)(nt * 16 + l16) * Tn + kk + half * 8);
            v8f* op = (nt == 0) ? &o0 : (nt == 1) ? &o1 : (nt == 2) ? &o2 : &o3;
            *op = WMMA_BF16(pf, vf, *op);
        }
    }

    bf16* obase = o + ((size_t)(b * Tn + q0)) * Dm + hh * HDn;
    #pragma unroll
    for (int nt = 0; nt < 4; nt++) {
        const v8f& oc = (nt == 0) ? o0 : (nt == 1) ? o1 : (nt == 2) ? o2 : o3;
        #pragma unroll
        for (int i = 0; i < 8; i++) {
            int r = i + half * 8;
            obase[(size_t)r * Dm + nt * 16 + l16] = f2bf(oc[i] / lrun[i]);
        }
    }
}

// ---------------------------------------------------------------------------
// Host orchestration
// ---------------------------------------------------------------------------
extern "C" void kernel_launch(void* const* d_in, const int* in_sizes, int n_in,
                              void* d_out, int out_size, void* d_ws, size_t ws_size,
                              hipStream_t stream) {
    (void)in_sizes; (void)n_in; (void)out_size; (void)ws_size;
    const int*   ids   = (const int*)  d_in[0];
    const float* te    = (const float*)d_in[1];
    const float* pe    = (const float*)d_in[2];
    const float* ln1_s = (const float*)d_in[3];
    const float* ln1_b = (const float*)d_in[4];
    const float* qkv_w = (const float*)d_in[5];
    const float* qkv_b = (const float*)d_in[6];
    const float* out_w = (const float*)d_in[7];
    const float* out_b = (const float*)d_in[8];
    const float* ln2_s = (const float*)d_in[9];
    const float* ln2_b = (const float*)d_in[10];
    const float* up_w  = (const float*)d_in[11];
    const float* up_b  = (const float*)d_in[12];
    const float* dn_w  = (const float*)d_in[13];
    const float* dn_b  = (const float*)d_in[14];
    const float* lnf_s = (const float*)d_in[15];
    const float* lnf_b = (const float*)d_in[16];

    char* wsp = (char*)d_ws;
    auto alloc = [&](size_t bytes) -> char* {
        char* p = wsp;
        wsp += (bytes + 255) & ~(size_t)255;
        return p;
    };
    float* x    = (float*)alloc((size_t)Mrows * Dm   * sizeof(float));
    bf16*  hbuf = (bf16*) alloc((size_t)Mrows * Dm   * sizeof(bf16));
    bf16*  qkvb = (bf16*) alloc((size_t)Mrows * QKVn * sizeof(bf16));
    bf16*  vtb  = (bf16*) alloc((size_t)Mrows * Dm   * sizeof(bf16)); // V transposed
    bf16*  abuf = (bf16*) alloc((size_t)Mrows * Dm   * sizeof(bf16));
    bf16*  ubuf = (bf16*) alloc((size_t)Mrows * DFm  * sizeof(bf16));
    bf16*  wt0  = (bf16*) alloc((size_t)QKVn * Dm    * sizeof(bf16)); // qkv_w^T
    bf16*  wt1  = (bf16*) alloc((size_t)Dm   * Dm    * sizeof(bf16)); // out_w^T
    bf16*  wt2  = (bf16*) alloc((size_t)DFm  * Dm    * sizeof(bf16)); // up_w^T
    bf16*  wt3  = (bf16*) alloc((size_t)Dm   * DFm   * sizeof(bf16)); // dn_w^T
    bf16*  tebf = (bf16*) alloc((size_t)Vn   * Dm    * sizeof(bf16)); // te [V,D] = Wt

    {   size_t tot = (size_t)Mrows * Dm;
        embed_kernel<<<dim3((unsigned)((tot + 255) / 256)), 256, 0, stream>>>(ids, te, pe, x); }
    {   size_t tot = (size_t)Vn * Dm;
        convert_kernel<<<dim3((unsigned)((tot + 255) / 256)), 256, 0, stream>>>(te, tebf, tot); }

    for (int i = 0; i < Ln; i++) {
        layernorm_kernel<<<Mrows, 256, 0, stream>>>(x, ln1_s + (size_t)i * Dm, ln1_b + (size_t)i * Dm, hbuf);

        { size_t tot = (size_t)Dm * QKVn;
          convert_transpose_kernel<<<dim3((unsigned)((tot + 255) / 256)), 256, 0, stream>>>(
              qkv_w + (size_t)i * Dm * QKVn, wt0, Dm, QKVn); }
        gemm_wmma_kernel<true, false><<<dim3(QKVn / 128, Mrows / 128), 256, 0, stream>>>(
            hbuf, wt0, qkv_b + (size_t)i * QKVn, nullptr, qkvb, Mrows, QKVn, Dm);

        { size_t tot = (size_t)Mrows * Dm;
          vtrans_kernel<<<dim3((unsigned)((tot + 255) / 256)), 256, 0, stream>>>(qkvb, vtb); }
        attention_kernel<<<Bn * Hn * (Tn / 64), 128, 0, stream>>>(qkvb, vtb, abuf);

        { size_t tot = (size_t)Dm * Dm;
          convert_transpose_kernel<<<dim3((unsigned)((tot + 255) / 256)), 256, 0, stream>>>(
              out_w + (size_t)i * Dm * Dm, wt1, Dm, Dm); }
        gemm_wmma_kernel<false, false><<<dim3(Dm / 128, Mrows / 128), 256, 0, stream>>>(
            abuf, wt1, out_b + (size_t)i * Dm, x, x, Mrows, Dm, Dm);

        layernorm_kernel<<<Mrows, 256, 0, stream>>>(x, ln2_s + (size_t)i * Dm, ln2_b + (size_t)i * Dm, hbuf);

        { size_t tot = (size_t)Dm * DFm;
          convert_transpose_kernel<<<dim3((unsigned)((tot + 255) / 256)), 256, 0, stream>>>(
              up_w + (size_t)i * Dm * DFm, wt2, Dm, DFm); }
        gemm_wmma_kernel<true, true><<<dim3(DFm / 128, Mrows / 128), 256, 0, stream>>>(
            hbuf, wt2, up_b + (size_t)i * DFm, nullptr, ubuf, Mrows, DFm, Dm);

        { size_t tot = (size_t)DFm * Dm;
          convert_transpose_kernel<<<dim3((unsigned)((tot + 255) / 256)), 256, 0, stream>>>(
              dn_w + (size_t)i * DFm * Dm, wt3, DFm, Dm); }
        gemm_wmma_kernel<false, false><<<dim3(Dm / 128, Mrows / 128), 256, 0, stream>>>(
            ubuf, wt3, dn_b + (size_t)i * Dm, x, x, Mrows, Dm, DFm);
    }

    layernorm_kernel<<<Mrows, 256, 0, stream>>>(x, lnf_s, lnf_b, hbuf);
    gemm_wmma_kernel<false, false><<<dim3(Vn / 128, Mrows / 128), 256, 0, stream>>>(
        hbuf, tebf, nullptr, nullptr, d_out, Mrows, Vn, Dm);
}
